// MultiHeadAttention_17420387352896
// MI455X (gfx1250) — matrix operand order repaired and checked by reading, not compiled, not run on hardware
//
#include <hip/hip_runtime.h>
#include <cstddef>

// ---------------------------------------------------------------- types
typedef __attribute__((ext_vector_type(16))) __bf16 v16bf;
typedef __attribute__((ext_vector_type(8)))  __bf16 v8bf;
typedef __attribute__((ext_vector_type(8)))  float  v8f;

#define DEV __device__ __forceinline__

static constexpr int  Bq = 8, Fq = 4, Hq = 8, Lq = 512, DM = 512, DH = 64;
static constexpr int  ROWS = Bq * Fq * Lq;                    // 16384
static constexpr size_t OUT_ELEMS = (size_t)ROWS * DM;        // 8,388,608
static constexpr size_t SC_ELEMS  = (size_t)Bq * Fq * Hq * Lq * Lq; // 67,108,864

// fp32 -> bf16, round-to-nearest-even
DEV unsigned short f2bf(float f) {
    unsigned int u = __float_as_uint(f);
    u += 0x7FFFu + ((u >> 16) & 1u);
    return (unsigned short)(u >> 16);
}
DEV __bf16 us2bf(unsigned short u) { return __builtin_bit_cast(__bf16, u); }

DEV v8f wmma_bf16(v16bf a, v16bf b, v8f c) {
    return __builtin_amdgcn_wmma_f32_16x16x32_bf16(
        /*neg_a=*/false, a, /*neg_b=*/false, b,
        /*c_mod=*/(short)0, c, /*reuse_a=*/false, /*reuse_b=*/false);
}

// A fragment (16x32 bf16, M x K) direct from global, row-major, 16B vector loads.
// ISA layout: lanes 0-15 hold row M=lane, K = 0..7 & 16..23 ; lanes 16-31: K = 8..15 & 24..31.
DEV v16bf load_a_g(const unsigned short* base, int stride, int lane) {
    const unsigned short* p = base + (size_t)(lane & 15) * stride + (lane >> 4) * 8;
    v8bf lo = __builtin_bit_cast(v8bf, *(const uint4*)(p));
    v8bf hi = __builtin_bit_cast(v8bf, *(const uint4*)(p + 16));
    return __builtin_shufflevector(lo, hi, 0,1,2,3,4,5,6,7,8,9,10,11,12,13,14,15);
}
// B fragment (32x16 bf16, K x N) where memory holds Bt row-major as [n][k]
// (i.e. B's column n is contiguous in k). lane -> col = lane&15, K = (lane>>4)*8 + 0..7 & 16 + (lane>>4)*8 + 0..7.
DEV v16bf load_b_g(const unsigned short* base, int stride, int lane) {
    const unsigned short* p = base + (size_t)(lane & 15) * stride + (lane >> 4) * 8;
    v8bf lo = __builtin_bit_cast(v8bf, *(const uint4*)(p));
    v8bf hi = __builtin_bit_cast(v8bf, *(const uint4*)(p + 16));
    return __builtin_shufflevector(lo, hi, 0,1,2,3,4,5,6,7,8,9,10,11,12,13,14,15);
}
// Same layouts but element-wise from (padded, unaligned-stride) LDS.
DEV v16bf load_a_lds(const unsigned short* base, int stride, int lane) {
    const unsigned short* p = base + (lane & 15) * stride + (lane >> 4) * 8;
    v16bf a;
#pragma unroll
    for (int i = 0; i < 8; ++i) { a[i] = us2bf(p[i]); a[8 + i] = us2bf(p[16 + i]); }
    return a;
}
DEV v16bf load_b_lds(const unsigned short* base, int stride, int lane) {
    const unsigned short* p = base + (lane & 15) * stride + (lane >> 4) * 8;
    v16bf b;
#pragma unroll
    for (int i = 0; i < 8; ++i) { b[i] = us2bf(p[i]); b[8 + i] = us2bf(p[16 + i]); }
    return b;
}

// ------------------------------------------------------- GEMM  C = A(f32) x B(f32)
// A: MxK row-major fp32, B: KxN row-major fp32, converted to bf16 while staging to LDS.
// Block tile 128x128x32, 256 threads = 8 waves in a 4(M) x 2(N) grid; 2x4 WMMA tiles/wave.
static constexpr int GBM = 128, GBN = 128, GBK = 32, GPAD = 2;

template <bool OUT_BF16>
__global__ void __launch_bounds__(256)
gemm_rc(const float* __restrict__ A, const float* __restrict__ B,
        void* __restrict__ Cout, int M, int N, int K) {
    __shared__ unsigned short As[GBM][GBK + GPAD];
    __shared__ unsigned short Bt[GBN][GBK + GPAD];   // B tile transposed: [n][k]
    const int tid = threadIdx.x, lane = tid & 31, w = tid >> 5;
    const int mb = blockIdx.y * GBM, nb = blockIdx.x * GBN;
    const int wm = (w & 3) * 32, wn = (w >> 2) * 64;

    v8f acc[2][4] = {};
    for (int kk = 0; kk < K; kk += GBK) {
        for (int i = tid; i < GBM * GBK; i += 256) {
            int r = i >> 5, c = i & 31;
            As[r][c] = f2bf(A[(size_t)(mb + r) * K + kk + c]);
        }
        for (int i = tid; i < GBK * GBN; i += 256) {
            int c = i & 127, r = i >> 7;
            Bt[c][r] = f2bf(B[(size_t)(kk + r) * N + nb + c]);
        }
        __syncthreads();
        v16bf af[2], bf[4];
#pragma unroll
        for (int mt = 0; mt < 2; ++mt) af[mt] = load_a_lds(&As[wm + mt * 16][0], GBK + GPAD, lane);
#pragma unroll
        for (int nt = 0; nt < 4; ++nt) bf[nt] = load_b_lds(&Bt[wn + nt * 16][0], GBK + GPAD, lane);
#pragma unroll
        for (int mt = 0; mt < 2; ++mt)
#pragma unroll
            for (int nt = 0; nt < 4; ++nt)
                acc[mt][nt] = wmma_bf16(af[mt], bf[nt], acc[mt][nt]);
        __syncthreads();
    }
    const int lrow = (lane >> 4) * 8, lcol = lane & 15;
#pragma unroll
    for (int mt = 0; mt < 2; ++mt)
#pragma unroll
        for (int nt = 0; nt < 4; ++nt)
#pragma unroll
            for (int v = 0; v < 8; ++v) {
                size_t row = (size_t)(mb + wm + mt * 16 + lrow + v);
                size_t col = (size_t)(nb + wn + nt * 16 + lcol);
                float val = acc[mt][nt][v];
                if (OUT_BF16) ((unsigned short*)Cout)[row * N + col] = f2bf(val);
                else          ((float*)Cout)[row * N + col] = val;
            }
}

// ------------------------------------------------ scores = Q Kt / 8 + res (+mask)
// grid: (n_tiles=4, m_tiles=4, bfh=256), block 256. A/B fragments straight from global bf16.
__global__ void __launch_bounds__(256)
scores_kernel(const unsigned short* __restrict__ Qp, const unsigned short* __restrict__ Kp,
              const float* __restrict__ res, const unsigned char* __restrict__ mask,
              float* __restrict__ Sout) {
    const int bfh = blockIdx.z, h = bfh & 7, bf = bfh >> 3;
    const int mb = blockIdx.y * 128, nb = blockIdx.x * 128;
    const int tid = threadIdx.x, lane = tid & 31, w = tid >> 5;
    const int wm = (w & 3) * 32, wn = (w >> 2) * 64;
    const unsigned short* Qb = Qp + (size_t)bf * Lq * DM + h * DH;
    const unsigned short* Kb = Kp + (size_t)bf * Lq * DM + h * DH;

    v8f acc[2][4] = {};
#pragma unroll
    for (int kk = 0; kk < DH; kk += 32) {
        v16bf af[2], bfr[4];
#pragma unroll
        for (int mt = 0; mt < 2; ++mt)
            af[mt] = load_a_g(Qb + (size_t)(mb + wm + mt * 16) * DM + kk, DM, lane);
#pragma unroll
        for (int nt = 0; nt < 4; ++nt)
            bfr[nt] = load_b_g(Kb + (size_t)(nb + wn + nt * 16) * DM + kk, DM, lane);
#pragma unroll
        for (int mt = 0; mt < 2; ++mt)
#pragma unroll
            for (int nt = 0; nt < 4; ++nt)
                acc[mt][nt] = wmma_bf16(af[mt], bfr[nt], acc[mt][nt]);
    }
    const int lrow = (lane >> 4) * 8, lcol = lane & 15;
    const size_t sb = (size_t)bfh * Lq * Lq;
#pragma unroll
    for (int mt = 0; mt < 2; ++mt)
#pragma unroll
        for (int nt = 0; nt < 4; ++nt)
#pragma unroll
            for (int v = 0; v < 8; ++v) {
                size_t row = (size_t)(mb + wm + mt * 16 + lrow + v);
                size_t col = (size_t)(nb + wn + nt * 16 + lcol);
                size_t idx = sb + row * Lq + col;
                float s = acc[mt][nt][v] * 0.125f + res[idx];
                if (mask[idx]) s = -1.0e9f;
                Sout[idx] = s;
            }
}

// --------------------------------- softmax over the QUERY axis (axis=3, faithful)
// One block = one (b,f,h) x 64-column stripe; whole 512x64 fp32 tile in dynamic LDS.
__global__ void __launch_bounds__(256)
softmax_q(const float* __restrict__ S, unsigned short* __restrict__ attn) {
    extern __shared__ float sm[];
    float* tile = sm;                 // [512][65]
    float* red  = sm + 512 * 65;      // [256]
    float* Mcol = red + 256;          // [64]
    float* Scol = Mcol + 64;          // [64]
    const int bfh = blockIdx.y, cb = blockIdx.x * 64, tid = threadIdx.x;
    const float* Sb = S + (size_t)bfh * Lq * Lq + cb;

    for (int i = tid; i < 512 * 64; i += 256) {
        int r = i >> 6, c = i & 63;
        tile[r * 65 + c] = Sb[(size_t)r * Lq + c];
    }
    __syncthreads();
    const int c = tid & 63, rq = tid >> 6;
    float m = -1.0e30f;
    for (int r = rq * 128; r < rq * 128 + 128; ++r) m = fmaxf(m, tile[r * 65 + c]);
    red[tid] = m;
    __syncthreads();
    if (rq == 0)
        Mcol[c] = fmaxf(fmaxf(red[c], red[64 + c]), fmaxf(red[128 + c], red[192 + c]));
    __syncthreads();
    const float mm = Mcol[c];
    float s = 0.f;
    for (int r = rq * 128; r < rq * 128 + 128; ++r) {
        float e = __expf(tile[r * 65 + c] - mm);
        tile[r * 65 + c] = e;
        s += e;
    }
    red[tid] = s;
    __syncthreads();
    if (rq == 0) Scol[c] = red[c] + red[64 + c] + red[128 + c] + red[192 + c];
    __syncthreads();
    const float inv = 1.0f / Scol[c];
    unsigned short* Ab = attn + (size_t)bfh * Lq * Lq + cb;
    for (int i = tid; i < 512 * 64; i += 256) {
        int r = i >> 6, cc = i & 63;      // cc == c for every iteration
        Ab[(size_t)r * Lq + cc] = f2bf(tile[r * 65 + cc] * inv);
    }
}

// -------------------------------------------- context = attn x V   (per b,f,h)
// V head tile (512x64) transposed into LDS -> B fragments contiguous in K.
__global__ void __launch_bounds__(256)
av_kernel(const unsigned short* __restrict__ attn, const unsigned short* __restrict__ Vp,
          float* __restrict__ ctx) {
    extern __shared__ unsigned short vt[];       // [64][512+8]
    const int VS = 512 + 8;
    const int bfh = blockIdx.y, h = bfh & 7, bf = bfh >> 3;
    const int mb = blockIdx.x * 128;
    const int tid = threadIdx.x, lane = tid & 31, w = tid >> 5;
    const unsigned short* Vb = Vp + (size_t)bf * Lq * DM + h * DH;

    for (int i = tid; i < 512 * 64; i += 256) {
        int k = i >> 6, d = i & 63;
        vt[d * VS + k] = Vb[(size_t)k * DM + d];
    }
    __syncthreads();

    const unsigned short* Ab = attn + (size_t)bfh * Lq * Lq;
    const int mrow = mb + w * 16;
    v8f acc[4] = {};
    for (int kk = 0; kk < Lq; kk += 32) {
        v16bf a = load_a_g(Ab + (size_t)mrow * Lq + kk, Lq, lane);
#pragma unroll
        for (int nt = 0; nt < 4; ++nt) {
            v16bf b = load_b_lds(vt + nt * 16 * VS + kk, VS, lane);
            acc[nt] = wmma_bf16(a, b, acc[nt]);
        }
    }
    const int lrow = (lane >> 4) * 8, lcol = lane & 15;
    float* Cb = ctx + (size_t)bf * Lq * DM + h * DH;
#pragma unroll
    for (int nt = 0; nt < 4; ++nt)
#pragma unroll
        for (int v = 0; v < 8; ++v)
            Cb[(size_t)(mrow + lrow + v) * DM + nt * 16 + lcol] = acc[nt][v];
}

// --------------------------------------------- LayerNorm(outbuf + input_Q)
__global__ void __launch_bounds__(256)
ln_kernel(const float* __restrict__ ob, const float* __restrict__ inq,
          const float* __restrict__ g, const float* __restrict__ bb,
          float* __restrict__ out) {
    __shared__ float r1[256], r2[256];
    const int row = blockIdx.x, tid = threadIdx.x;
    const float* o = ob + (size_t)row * DM;
    const float* q = inq + (size_t)row * DM;
    float x0 = o[tid] + q[tid];
    float x1 = o[tid + 256] + q[tid + 256];
    r1[tid] = x0 + x1;
    r2[tid] = x0 * x0 + x1 * x1;
    __syncthreads();
    for (int s = 128; s > 0; s >>= 1) {
        if (tid < s) { r1[tid] += r1[tid + s]; r2[tid] += r2[tid + s]; }
        __syncthreads();
    }
    const float mu = r1[0] * (1.0f / DM);
    const float var = r2[0] * (1.0f / DM) - mu * mu;
    const float rs = rsqrtf(var + 1e-5f);
    float* dst = out + (size_t)row * DM;
    dst[tid]       = (x0 - mu) * rs * g[tid] + bb[tid];
    dst[tid + 256] = (x1 - mu) * rs * g[tid + 256] + bb[tid + 256];
}

// ---------------------------------------------------------------- launcher
extern "C" void kernel_launch(void* const* d_in, const int* in_sizes, int n_in,
                              void* d_out, int out_size, void* d_ws, size_t ws_size,
                              hipStream_t stream) {
    (void)in_sizes; (void)n_in; (void)out_size; (void)ws_size;
    const float*         inQ  = (const float*)d_in[0];
    const float*         inK  = (const float*)d_in[1];
    const float*         inV  = (const float*)d_in[2];
    const unsigned char* mask = (const unsigned char*)d_in[3];
    const float*         res  = (const float*)d_in[4];
    const float*         Wq   = (const float*)d_in[5];
    const float*         Wk   = (const float*)d_in[6];
    const float*         Wv   = (const float*)d_in[7];
    const float*         Wfc  = (const float*)d_in[8];
    const float*         lng  = (const float*)d_in[9];
    const float*         lnb  = (const float*)d_in[10];

    float* out_final  = (float*)d_out;                  // [16384,512]
    float* out_scores = (float*)d_out + OUT_ELEMS;      // [256,512,512]

    char* ws = (char*)d_ws;
    const size_t PROJ_BF16 = (size_t)ROWS * DM * 2;     // 16 MiB
    unsigned short* Qp   = (unsigned short*)(ws);
    unsigned short* Kp   = (unsigned short*)(ws + PROJ_BF16);
    unsigned short* Vp   = (unsigned short*)(ws + 2 * PROJ_BF16);
    unsigned short* attn = (unsigned short*)(ws + 3 * PROJ_BF16);          // 128 MiB
    float*          ctx  = (float*)(ws + 3 * PROJ_BF16 + SC_ELEMS * 2);    // 32 MiB
    float*          outb = (float*)(ws + 3 * PROJ_BF16 + SC_ELEMS * 2 + OUT_ELEMS * 4);

    const dim3 gGemm(DM / GBN, ROWS / GBM);             // (4, 128)
    gemm_rc<true ><<<gGemm, 256, 0, stream>>>(inQ, Wq, Qp,  ROWS, DM, DM);
    gemm_rc<true ><<<gGemm, 256, 0, stream>>>(inK, Wk, Kp,  ROWS, DM, DM);
    gemm_rc<true ><<<gGemm, 256, 0, stream>>>(inV, Wv, Vp,  ROWS, DM, DM);

    scores_kernel<<<dim3(4, 4, Bq * Fq * Hq), 256, 0, stream>>>(Qp, Kp, res, mask, out_scores);

    const size_t SOFT_SMEM = (size_t)(512 * 65 + 256 + 64 + 64) * sizeof(float); // ~134 KB (of 320 KB LDS)
    softmax_q<<<dim3(Lq / 64, Bq * Fq * Hq), 256, SOFT_SMEM, stream>>>(out_scores, attn);

    const size_t AV_SMEM = (size_t)64 * (512 + 8) * sizeof(unsigned short);      // ~65 KB
    av_kernel<<<dim3(Lq / 128, Bq * Fq * Hq), 256, AV_SMEM, stream>>>(attn, Vp, ctx);

    gemm_rc<false><<<gGemm, 256, 0, stream>>>(ctx, Wfc, outb, ROWS, DM, DM);

    ln_kernel<<<ROWS, 256, 0, stream>>>(outb, inQ, lng, lnb, out_final);
}
